// MoEResidualAttentionBlock_52364241273505
// MI455X (gfx1250) — matrix operand order repaired and checked by reading, not compiled, not run on hardware
//
#include <hip/hip_runtime.h>
#include <hip/hip_bf16.h>

// ---------------- problem constants ----------------
constexpr int S_   = 512;
constexpr int B_   = 16;
constexpr int D_   = 768;
constexpr int H_   = 12;
constexpr int E_   = 8;
constexpr int KTOP = 2;
constexpr int DF_  = 3072;
constexpr int HD_  = 64;
constexpr int T_   = S_ * B_;   // 8192 tokens
constexpr int D3_  = 3 * D_;    // 2304

typedef _Float16 half_t;
typedef __attribute__((ext_vector_type(8)))  _Float16 h8;
typedef __attribute__((ext_vector_type(16))) _Float16 h16v;
typedef __attribute__((ext_vector_type(8)))  float    f8;
typedef __attribute__((ext_vector_type(4)))  unsigned int u32x4;
typedef __attribute__((ext_vector_type(4)))  int      i32x4;
typedef __attribute__((ext_vector_type(8)))  int      i32x8;

#define WMMA_F16(a, b, c) \
  __builtin_amdgcn_wmma_f32_16x16x32_f16(false, (a), false, (b), (short)0, (c), false, false)

#if __has_builtin(__builtin_amdgcn_tensor_load_to_lds)
#define HAVE_TDM 1
#else
#define HAVE_TDM 0
#endif

__device__ inline h16v make_h16(h8 lo, h8 hh) {
  return __builtin_shufflevector(lo, hh, 0, 1, 2, 3, 4, 5, 6, 7, 8, 9, 10, 11, 12, 13, 14, 15);
}

// A-fragment (16x32 f16): lane (hi = lane>>4) holds row M = lane&15 with
// K-halves [hi*8 .. hi*8+7] and [16+hi*8 .. 16+hi*8+7] (ISA 7.12.2 table).
// `rowbase` points at (row start + k-slice start) in a row-major LDS tile.
__device__ inline h16v frag_a(const half_t* rowbase, int hi) {
  h8 lo = *(const h8*)(rowbase + hi * 8);
  h8 hh = *(const h8*)(rowbase + 16 + hi * 8);
  return make_h16(lo, hh);
}

// B-fragment (32x16 f16) where the B operand's column n is a contiguous
// row of the staged tile (W row-major [N,K], or a pre-transposed V tile):
// lane holds K-halves [hi*8 .. hi*8+7] and [16+hi*8 .. 16+hi*8+7] (same as A).
__device__ inline h16v frag_b_rows(const half_t* rowbase, int hi) {
  h8 lo = *(const h8*)(rowbase + hi * 8);
  h8 hh = *(const h8*)(rowbase + 16 + hi * 8);
  return make_h16(lo, hh);
}

// ---------------- Tensor Data Mover: 2-D tile global -> LDS ----------------
// Descriptor per CDNA5 ISA. One call = whole tile,
// tracked on TENSORcnt. Issue from a single wave; EXEC is ignored.
__device__ inline void tdm_load_2d_tile(unsigned lds_addr, const void* gptr,
                                        unsigned tile_w_elems, unsigned tile_h,
                                        unsigned row_stride_elems) {
#if HAVE_TDM
  unsigned long long ga = (unsigned long long)gptr;
  u32x4 g0;
  g0[0] = 1u;                                       // count=1, user descriptor
  g0[1] = lds_addr;                                 // LDS byte address
  g0[2] = (unsigned)(ga & 0xffffffffu);             // global_addr[31:0]
  g0[3] = (unsigned)((ga >> 32) & 0x01ffffffu)      // global_addr[56:32]
          | (2u << 30);                             // type = 2 ("image")
  i32x8 g1;
  g1[0] = (int)(1u << 16);                          // data_size=1 (2 bytes)
  g1[1] = (int)(tile_w_elems << 16);                // tensor_dim0[15:0]
  g1[2] = (int)(tile_h << 16);                      // tensor_dim1[15:0]
  g1[3] = (int)(tile_w_elems << 16);                // tile_dim0
  g1[4] = (int)tile_h;                              // tile_dim1 (tile_dim2=0)
  g1[5] = (int)row_stride_elems;                    // tensor_dim0_stride[31:0]
  g1[6] = 0;                                        // stride hi / dim1_stride lo
  g1[7] = 0;
  i32x4 gz = {0, 0, 0, 0};
#if __clang_major__ >= 23
  i32x8 gz8 = {0, 0, 0, 0, 0, 0, 0, 0};
  __builtin_amdgcn_tensor_load_to_lds(g0, g1, gz, gz, gz8, 0);
#else
  __builtin_amdgcn_tensor_load_to_lds(g0, g1, gz, gz, 0);
#endif
#endif
}

__device__ inline void tdm_wait0() {
#if __has_builtin(__builtin_amdgcn_s_wait_tensorcnt)
  __builtin_amdgcn_s_wait_tensorcnt(0);
#endif
}

// ---------------- small utility kernels ----------------
__global__ void cvt_f32_f16_kernel(const float* __restrict__ src,
                                   half_t* __restrict__ dst, long n) {
  long i = (long)blockIdx.x * blockDim.x + threadIdx.x;
  long stride = (long)gridDim.x * blockDim.x;
  for (; i < n; i += stride) dst[i] = (half_t)src[i];
}

__global__ void zero_cnt_kernel(int* cnt) {
  if (threadIdx.x < E_) cnt[threadIdx.x] = 0;
}

// LayerNorm over D_=768, one block (256 threads) per token, f16 output.
__global__ __launch_bounds__(256) void ln_cast_kernel(
    const float* __restrict__ x, const float* __restrict__ g,
    const float* __restrict__ b, half_t* __restrict__ out) {
  const int t = blockIdx.x;
  const float* xr = x + (long)t * D_;
  __shared__ float red[256];

  float loc = 0.f;
  for (int d = threadIdx.x; d < D_; d += 256) loc += xr[d];
  red[threadIdx.x] = loc;
  __syncthreads();
  for (int s = 128; s > 0; s >>= 1) {
    if (threadIdx.x < s) red[threadIdx.x] += red[threadIdx.x + s];
    __syncthreads();
  }
  const float mean = red[0] / (float)D_;
  __syncthreads();

  loc = 0.f;
  for (int d = threadIdx.x; d < D_; d += 256) {
    float dv = xr[d] - mean;
    loc += dv * dv;
  }
  red[threadIdx.x] = loc;
  __syncthreads();
  for (int s = 128; s > 0; s >>= 1) {
    if (threadIdx.x < s) red[threadIdx.x] += red[threadIdx.x + s];
    __syncthreads();
  }
  const float rstd = rsqrtf(red[0] / (float)D_ + 1e-5f);
  __syncthreads();

  for (int d = threadIdx.x; d < D_; d += 256)
    out[(long)t * D_ + d] = (half_t)(((xr[d] - mean) * rstd) * g[d] + b[d]);
}

// ---------------- WMMA GEMM: C[M,N] = A[M,K] @ W[N,K]^T (+ epilogue) ----------------
// EPI 0: + bias, store f16 (QKV)
// EPI 1: + bias + residual, store f32 to two destinations (out-proj)
// EPI 2: + bias, quickGELU, store f16; A rows gathered via row_idx (FFN up)
// EPI 3: + bias, scale by row_scale, scatter-accumulate f32 via row_idx (FFN down)
template <int EPI>
__global__ __launch_bounds__(256) void gemm_wmma_kernel(
    const half_t* __restrict__ A, const half_t* __restrict__ W,
    const float* __restrict__ bias, const float* __restrict__ resid,
    const int* __restrict__ row_idx, const float* __restrict__ row_scale,
    const int* __restrict__ cntp, int Mmax, int N, int Kd,
    float* __restrict__ Cf, half_t* __restrict__ Ch, float* __restrict__ Cf2,
    int ldc) {
  int M = Mmax;
  if (cntp) { M = *cntp; if (M > Mmax) M = Mmax; }
  const int m0 = blockIdx.y * 128;
  const int n0 = blockIdx.x * 128;
  if (m0 >= M) return;

  __shared__ half_t As[128 * 32];
  __shared__ half_t Bs[128 * 32];

  const int tid  = threadIdx.x;
  const int lane = tid & 31;
  const int wv   = tid >> 5;      // wave 0..7
  const int hi   = lane >> 4;     // lane half
  const int ln15 = lane & 15;
  const int wr   = wv >> 2;       // wave row 0..1 (64-row bands)
  const int wc   = wv & 3;        // wave col 0..3 (32-col bands)

  f8 acc[8] = {};

  const int lr = tid >> 1;        // 0..127: tile row this thread stages
  const int lc = (tid & 1) * 16;  // 0 or 16: 16-half segment

  for (int kk = 0; kk < Kd; kk += 32) {
    {
      const int row = m0 + lr;
      if (row < M) {
        long ar = row;
        if (EPI == 2) ar = row_idx[row];  // gathered activations
        const half_t* src = A + ar * (long)Kd + kk + lc;
        *(h8*)(As + lr * 32 + lc)     = *(const h8*)src;
        *(h8*)(As + lr * 32 + lc + 8) = *(const h8*)(src + 8);
        if (kk + 32 < Kd) __builtin_prefetch(src + 32, 0, 1);
      } else {
        h8 z;
#pragma unroll
        for (int i = 0; i < 8; ++i) z[i] = (_Float16)0.f;
        *(h8*)(As + lr * 32 + lc)     = z;
        *(h8*)(As + lr * 32 + lc + 8) = z;
      }
      const half_t* wsrc = W + (long)(n0 + lr) * Kd + kk + lc;
      *(h8*)(Bs + lr * 32 + lc)     = *(const h8*)wsrc;
      *(h8*)(Bs + lr * 32 + lc + 8) = *(const h8*)(wsrc + 8);
      if (kk + 32 < Kd) __builtin_prefetch(wsrc + 32, 0, 1);
    }
    __syncthreads();
#pragma unroll
    for (int str = 0; str < 4; ++str) {
      h16v af = frag_a(As + (wr * 64 + str * 16 + ln15) * 32, hi);
#pragma unroll
      for (int stc = 0; stc < 2; ++stc) {
        h16v bf = frag_b_rows(Bs + (wc * 32 + stc * 16 + ln15) * 32, hi);
        acc[str * 2 + stc] = WMMA_F16(af, bf, acc[str * 2 + stc]);
      }
    }
    __syncthreads();
  }

#pragma unroll
  for (int st = 0; st < 8; ++st) {
    const int str = st >> 1, stc = st & 1;
    const int col = n0 + wc * 32 + stc * 16 + ln15;
    const float bb = bias ? bias[col] : 0.f;
#pragma unroll
    for (int r = 0; r < 8; ++r) {
      const int row = m0 + wr * 64 + str * 16 + 8 * hi + r;
      if (row >= M) continue;
      float v = acc[st][r] + bb;
      if (EPI == 0) {
        Ch[(long)row * ldc + col] = (half_t)v;
      } else if (EPI == 1) {
        float o = v + resid[(long)row * ldc + col];
        Cf[(long)row * ldc + col]  = o;
        Cf2[(long)row * ldc + col] = o;
      } else if (EPI == 2) {
        float gel = v / (1.f + __expf(-1.702f * v));  // x * sigmoid(1.702 x)
        Ch[(long)row * ldc + col] = (half_t)gel;
      } else {  // EPI == 3
        const int   tok = row_idx[row];
        const float wgt = row_scale[row];
        Cf[(long)tok * ldc + col] += v * wgt;
      }
    }
  }
}

// ---------------- flash attention ----------------
// grid = (S/64, B*H); block 256 (8 waves). q-block of 64 rows, stream over
// 64-row K/V blocks with online softmax. HD = 64.
// K tile staged via Tensor Data Mover (async, TENSORcnt); V staged manually
// *transposed* so the P@V B-fragment is two contiguous ds_load_b128.
constexpr int VT_LD = 72;  // padded row stride (halves) for transposed V

__global__ __launch_bounds__(256) void flash_attn_kernel(
    const half_t* __restrict__ qkv, half_t* __restrict__ attn_out) {
  const int qb    = blockIdx.x;       // 0..7
  const int bh    = blockIdx.y;       // 0..191
  const int batch = bh / H_;
  const int head  = bh % H_;

  __shared__ half_t Qs[64 * 64];
  __shared__ half_t Ks[64 * 64];
  __shared__ half_t Vt[64 * VT_LD];   // transposed V: Vt[col][row]
  __shared__ half_t Ps[64 * 64];
  __shared__ float  Ss[64 * 64];
  __shared__ float  mrow[64], lrow[64], arow[64];

  const int tid  = threadIdx.x;
  const int lane = tid & 31;
  const int wv   = tid >> 5;
  const int hi   = lane >> 4;
  const int ln15 = lane & 15;

  // load+scale Q tile (scale = 1/sqrt(64) = 0.125)
  {
    const int r = tid >> 2;
    const int c = (tid & 3) * 16;
    const int s = qb * 64 + r;
    const half_t* src = qkv + ((long)(s * B_ + batch)) * D3_ + head * HD_ + c;
#pragma unroll
    for (int i = 0; i < 16; ++i)
      Qs[r * 64 + c + i] = (half_t)((float)src[i] * 0.125f);
  }
  if (tid < 64) { mrow[tid] = -1e30f; lrow[tid] = 0.f; }

  f8 Oacc[2] = {};
  const int t0 = wv * 2;  // 2 of the 16 (4x4) 16x16 output tiles per wave
  __syncthreads();

  for (int kb = 0; kb < S_ / 64; ++kb) {
    // ---- stage V transposed (all threads) ----
    {
      const int r = tid >> 2;
      const int c = (tid & 3) * 16;
      const int s = kb * 64 + r;
      const half_t* vp = qkv + ((long)(s * B_ + batch)) * D3_ + 2 * D_ + head * HD_ + c;
      h8 v0 = *(const h8*)vp;
      h8 v1 = *(const h8*)(vp + 8);
#pragma unroll
      for (int i = 0; i < 8; ++i) Vt[(c + i) * VT_LD + r] = v0[i];
#pragma unroll
      for (int i = 0; i < 8; ++i) Vt[(c + 8 + i) * VT_LD + r] = v1[i];
    }
    // ---- stage K tile: Tensor Data Mover (wave 0), fallback manual ----
#if HAVE_TDM
    if (wv == 0) {
      const half_t* kbase =
          qkv + ((long)((kb * 64) * B_ + batch)) * D3_ + D_ + head * HD_;
      tdm_load_2d_tile((unsigned)(unsigned long long)(&Ks[0]), kbase,
                       /*tile_w=*/64, /*tile_h=*/64,
                       /*row_stride=*/(unsigned)(B_ * D3_));
      tdm_wait0();  // issuing wave drains TENSORcnt before the barrier
    }
#else
    {
      const int r = tid >> 2;
      const int c = (tid & 3) * 16;
      const int s = kb * 64 + r;
      const half_t* kp = qkv + ((long)(s * B_ + batch)) * D3_ + D_ + head * HD_ + c;
      *(h8*)(Ks + r * 64 + c)     = *(const h8*)kp;
      *(h8*)(Ks + r * 64 + c + 8) = *(const h8*)(kp + 8);
    }
#endif
    __syncthreads();

    // ---- scores S = Q * K^T (64x64), two k-slices of 32 ----
#pragma unroll
    for (int ti = 0; ti < 2; ++ti) {
      const int t = t0 + ti, rb = t >> 2, cb = t & 3;
      f8 sacc = {};
#pragma unroll
      for (int ks = 0; ks < 2; ++ks) {
        h16v af = frag_a(Qs + (rb * 16 + ln15) * 64 + ks * 32, hi);
        h16v bf = frag_b_rows(Ks + (cb * 16 + ln15) * 64 + ks * 32, hi);
        sacc = WMMA_F16(af, bf, sacc);
      }
#pragma unroll
      for (int r = 0; r < 8; ++r)
        Ss[(rb * 16 + r + 8 * hi) * 64 + cb * 16 + ln15] = sacc[r];
    }
    __syncthreads();

    // ---- online softmax, one thread per q-row ----
    if (tid < 64) {
      const float m_old = mrow[tid];
      float mx = m_old;
      for (int j = 0; j < 64; ++j) mx = fmaxf(mx, Ss[tid * 64 + j]);
      const float alpha = __expf(m_old - mx);
      float ssum = 0.f;
      for (int j = 0; j < 64; ++j) {
        float p = __expf(Ss[tid * 64 + j] - mx);
        Ps[tid * 64 + j] = (half_t)p;
        ssum += p;
      }
      mrow[tid] = mx;
      lrow[tid] = lrow[tid] * alpha + ssum;
      arow[tid] = alpha;
    }
    __syncthreads();

    // ---- O = O*alpha + P @ V (B-fragment from transposed V: contiguous) ----
#pragma unroll
    for (int ti = 0; ti < 2; ++ti) {
      const int t = t0 + ti, rb = t >> 2, cb = t & 3;
      f8 o = Oacc[ti];
#pragma unroll
      for (int r = 0; r < 8; ++r) o[r] *= arow[rb * 16 + r + 8 * hi];
#pragma unroll
      for (int ks = 0; ks < 2; ++ks) {
        h16v af = frag_a(Ps + (rb * 16 + ln15) * 64 + ks * 32, hi);
        h16v bf = frag_b_rows(Vt + (cb * 16 + ln15) * VT_LD + ks * 32, hi);
        o = WMMA_F16(af, bf, o);
      }
      Oacc[ti] = o;
    }
    __syncthreads();
  }

  // ---- normalize and write [S,B,D] (head-sliced) ----
#pragma unroll
  for (int ti = 0; ti < 2; ++ti) {
    const int t = t0 + ti, rb = t >> 2, cb = t & 3;
#pragma unroll
    for (int r = 0; r < 8; ++r) {
      const int row = rb * 16 + r + 8 * hi;
      const int s   = qb * 64 + row;
      const float inv = 1.f / lrow[row];
      attn_out[((long)(s * B_ + batch)) * D_ + head * HD_ + cb * 16 + ln15] =
          (half_t)(Oacc[ti][r] * inv);
    }
  }
}

// ---------------- router: logits, softmax, top-2, expert binning ----------------
__global__ __launch_bounds__(256) void router_kernel(
    const half_t* __restrict__ h2, const float* __restrict__ gate_w,
    float* __restrict__ logits_out, float* __restrict__ rw_out,
    float* __restrict__ sel_out, int* __restrict__ expert_cnt,
    int* __restrict__ expert_idx, float* __restrict__ expert_wt) {
  const int t = blockIdx.x * blockDim.x + threadIdx.x;
  if (t >= T_) return;

  float lg[E_];
#pragma unroll
  for (int e = 0; e < E_; ++e) lg[e] = 0.f;
  for (int d = 0; d < D_; ++d) {
    const float hv = (float)h2[(long)t * D_ + d];
#pragma unroll
    for (int e = 0; e < E_; ++e) lg[e] += hv * gate_w[e * D_ + d];
  }
  float mx = lg[0];
#pragma unroll
  for (int e = 1; e < E_; ++e) mx = fmaxf(mx, lg[e]);
  float p[E_], psum = 0.f;
#pragma unroll
  for (int e = 0; e < E_; ++e) { p[e] = __expf(lg[e] - mx); psum += p[e]; }
#pragma unroll
  for (int e = 0; e < E_; ++e) {
    p[e] /= psum;
    logits_out[(long)t * E_ + e] = lg[e];
  }
  int e0 = 0;
#pragma unroll
  for (int e = 1; e < E_; ++e) if (p[e] > p[e0]) e0 = e;
  int e1 = (e0 == 0) ? 1 : 0;
#pragma unroll
  for (int e = 0; e < E_; ++e) if (e != e0 && p[e] > p[e1]) e1 = e;

  const float s2 = p[e0] + p[e1];
  const float w0 = p[e0] / s2, w1 = p[e1] / s2;
  rw_out[t * KTOP + 0]  = w0;
  rw_out[t * KTOP + 1]  = w1;
  sel_out[t * KTOP + 0] = (float)e0;
  sel_out[t * KTOP + 1] = (float)e1;

  int pos = atomicAdd(&expert_cnt[e0], 1);
  expert_idx[(long)e0 * T_ + pos] = t;
  expert_wt[(long)e0 * T_ + pos]  = w0;
  pos = atomicAdd(&expert_cnt[e1], 1);
  expert_idx[(long)e1 * T_ + pos] = t;
  expert_wt[(long)e1 * T_ + pos]  = w1;
}

// ---------------- host orchestration ----------------
extern "C" void kernel_launch(void* const* d_in, const int* in_sizes, int n_in,
                              void* d_out, int out_size, void* d_ws, size_t ws_size,
                              hipStream_t stream) {
  const float* x          = (const float*)d_in[0];
  const float* ln1_g      = (const float*)d_in[1];
  const float* ln1_b      = (const float*)d_in[2];
  const float* in_proj_w  = (const float*)d_in[3];
  const float* in_proj_b  = (const float*)d_in[4];
  const float* out_proj_w = (const float*)d_in[5];
  const float* out_proj_b = (const float*)d_in[6];
  const float* ln2_g      = (const float*)d_in[7];
  const float* ln2_b      = (const float*)d_in[8];
  const float* gate_w     = (const float*)d_in[9];
  const float* w1         = (const float*)d_in[10];
  const float* b1         = (const float*)d_in[11];
  const float* w2         = (const float*)d_in[12];
  const float* b2         = (const float*)d_in[13];
  (void)in_sizes; (void)n_in; (void)out_size; (void)ws_size;

  // output layout: [output T*D][router_logits T*E][routing_weights T*K][selected T*K]
  float* out_x      = (float*)d_out;
  float* out_logits = out_x + (long)T_ * D_;
  float* out_rw     = out_logits + (long)T_ * E_;
  float* out_sel    = out_rw + (long)T_ * KTOP;

  // workspace carve (~232 MB)
  size_t off = 0;
  auto carve = [&](size_t bytes) -> void* {
    void* p = (char*)d_ws + off;
    off = (off + bytes + 255) & ~(size_t)255;
    return p;
  };
  half_t* h16    = (half_t*)carve((size_t)T_ * D_ * 2);        // LN1 out
  half_t* wqkvh  = (half_t*)carve((size_t)D3_ * D_ * 2);       // in_proj f16
  half_t* wouth  = (half_t*)carve((size_t)D_ * D_ * 2);        // out_proj f16
  half_t* w1h    = (half_t*)carve((size_t)E_ * DF_ * D_ * 2);  // w1 f16
  half_t* w2h    = (half_t*)carve((size_t)E_ * D_ * DF_ * 2);  // w2 f16
  half_t* qkv16  = (half_t*)carve((size_t)T_ * D3_ * 2);       // QKV
  half_t* attn16 = (half_t*)carve((size_t)T_ * D_ * 2);        // attention out
  float*  x_attn = (float*)carve((size_t)T_ * D_ * 4);         // residual stream
  half_t* h2_16  = (half_t*)carve((size_t)T_ * D_ * 2);        // LN2 out
  half_t* act16  = (half_t*)carve((size_t)T_ * DF_ * 2);       // FFN intermediate
  int*    eidx   = (int*)carve((size_t)E_ * T_ * 4);
  float*  ewt    = (float*)carve((size_t)E_ * T_ * 4);
  int*    ecnt   = (int*)carve((size_t)E_ * 4);

  // 0) zero expert counters
  zero_cnt_kernel<<<1, 32, 0, stream>>>(ecnt);

  // 1) weight conversions f32 -> f16
  cvt_f32_f16_kernel<<<512, 256, 0, stream>>>(in_proj_w, wqkvh, (long)D3_ * D_);
  cvt_f32_f16_kernel<<<512, 256, 0, stream>>>(out_proj_w, wouth, (long)D_ * D_);
  cvt_f32_f16_kernel<<<1024, 256, 0, stream>>>(w1, w1h, (long)E_ * DF_ * D_);
  cvt_f32_f16_kernel<<<1024, 256, 0, stream>>>(w2, w2h, (long)E_ * D_ * DF_);

  // 2) LN1
  ln_cast_kernel<<<T_, 256, 0, stream>>>(x, ln1_g, ln1_b, h16);

  // 3) QKV GEMM: [T,768] x [2304,768]^T + b -> qkv16
  gemm_wmma_kernel<0><<<dim3(D3_ / 128, T_ / 128), 256, 0, stream>>>(
      h16, wqkvh, in_proj_b, nullptr, nullptr, nullptr, nullptr,
      T_, D3_, D_, nullptr, qkv16, nullptr, D3_);

  // 4) flash attention
  flash_attn_kernel<<<dim3(S_ / 64, B_ * H_), 256, 0, stream>>>(qkv16, attn16);

  // 5) out-proj GEMM + residual -> x_attn (ws) and d_out
  gemm_wmma_kernel<1><<<dim3(D_ / 128, T_ / 128), 256, 0, stream>>>(
      attn16, wouth, out_proj_b, x, nullptr, nullptr, nullptr,
      T_, D_, D_, x_attn, nullptr, out_x, D_);

  // 6) LN2
  ln_cast_kernel<<<T_, 256, 0, stream>>>(x_attn, ln2_g, ln2_b, h2_16);

  // 7) router + top-2 binning
  router_kernel<<<T_ / 256, 256, 0, stream>>>(
      h2_16, gate_w, out_logits, out_rw, out_sel, ecnt, eidx, ewt);

  // 8) grouped MoE GEMMs (per expert; dynamic M via ecnt[e], blocks early-exit)
  for (int e = 0; e < E_; ++e) {
    gemm_wmma_kernel<2><<<dim3(DF_ / 128, T_ / 128), 256, 0, stream>>>(
        h2_16, w1h + (long)e * DF_ * D_, b1 + (long)e * DF_, nullptr,
        eidx + (long)e * T_, nullptr, ecnt + e,
        T_, DF_, D_, nullptr, act16, nullptr, DF_);
    gemm_wmma_kernel<3><<<dim3(D_ / 128, T_ / 128), 256, 0, stream>>>(
        act16, w2h + (long)e * D_ * DF_, b2 + (long)e * D_, nullptr,
        eidx + (long)e * T_, ewt + (long)e * T_, ecnt + e,
        T_, D_, DF_, out_x, nullptr, nullptr, D_);
  }
}
